// GeneratorBlock_46213848105892
// MI455X (gfx1250) — matrix operand order repaired and checked by reading, not compiled, not run on hardware
//
#include <hip/hip_runtime.h>
#include <math.h>

// ---------- types for WMMA ----------
typedef __attribute__((ext_vector_type(16))) __bf16        v16bf;
typedef __attribute__((ext_vector_type(8)))  float         v8f;
typedef __attribute__((ext_vector_type(8)))  unsigned int  v8u;

// exact pointee type expected by the async-LDS builtin (from compiler diagnostic):
//   int __attribute__((vector_size(16)))  in AS(1) / AS(3)
typedef int v4i __attribute__((vector_size(16)));
typedef __attribute__((address_space(1))) v4i* gv4i_p;
typedef __attribute__((address_space(3))) v4i* lv4i_p;

#if defined(__has_builtin)
#if __has_builtin(__builtin_amdgcn_global_load_async_to_lds_b128)
#define HAVE_ASYNC_LDS 1
#endif
#endif

__device__ __forceinline__ void wait_async0() {
#if defined(HAVE_ASYNC_LDS)
#if __has_builtin(__builtin_amdgcn_s_wait_asynccnt)
    __builtin_amdgcn_s_wait_asynccnt(0);
#else
    asm volatile("s_wait_asynccnt 0x0" ::: "memory");
#endif
#endif
}

__device__ __forceinline__ unsigned short f2bf(float f) {
    unsigned int u = __float_as_uint(f);
    unsigned int r = (u + 0x7FFFu + ((u >> 16) & 1u)) >> 16;
    return (unsigned short)r;
}

// ---------- prep: f32 -> bf16 (activations) ----------
__global__ __launch_bounds__(256) void cvt_bf16_kernel(const float* __restrict__ in,
                                                       unsigned short* __restrict__ out,
                                                       int n, float scale) {
    int i = blockIdx.x * 256 + threadIdx.x;
    if (i < n) out[i] = f2bf(in[i] * scale);
}

// ---------- prep: weights f32 -> bf16, prepacked in WMMA B-fragment chunk order ----
// Chunk c covers K rows [32c, 32c+32). Within a chunk (4096 ushorts = 8KB):
//   element (kk, n) -> ushort index (kkp*128 + col)*2 + (kk&1),
//   kkp = kk>>1, col = (n + (kkp>=8 ? 32 : 0)) & 127   (bank swizzle)
__global__ __launch_bounds__(256) void cvt_pack_w_kernel(const float* __restrict__ in, // [K,128]
                                                         unsigned short* __restrict__ out,
                                                         int total, float scale) {
    int i = blockIdx.x * 256 + threadIdx.x;
    if (i >= total) return;
    int k = i >> 7, n = i & 127;
    int chunk = k >> 5, kk = k & 31, kkp = kk >> 1;
    int col = (n + ((kkp & 8) ? 32 : 0)) & 127;
    size_t dst = (size_t)chunk * 4096 + (size_t)(kkp * 128 + col) * 2 + (kk & 1);
    out[dst] = f2bf(in[i] * scale);
}

// ---------- style dense: dlatent[512] @ (sw/sqrt(512)) + sb -> [B, 2C] ----------
__global__ __launch_bounds__(256) void style_kernel(const float* __restrict__ styles, // [8,2,512]
                                                    const float* __restrict__ sw1,
                                                    const float* __restrict__ sb1,
                                                    const float* __restrict__ sw2,
                                                    const float* __restrict__ sb2,
                                                    float* __restrict__ st1,          // [8,256]
                                                    float* __restrict__ st2) {
    int b = blockIdx.x, layer = blockIdx.y, j = threadIdx.x;
    const float* sv = styles + (size_t)(b * 2 + layer) * 512;
    const float* sw = layer ? sw2 : sw1;
    const float* sb = layer ? sb2 : sb1;
    float acc = 0.f;
    for (int k = 0; k < 512; ++k) acc += sv[k] * sw[k * 256 + j];
    float r = acc * 0.04419417382415922f /* 1/sqrt(512) */ + sb[j];
    (layer ? st2 : st1)[b * 256 + j] = r;
}

// ---------- implicit-GEMM 3x3 conv with WMMA bf16, double-buffered LDS ----------
// Output tile per WG: 16x8 spatial (M=128) x 128 channels (N=128). 8 waves.
// Each wave: 2 M-subtiles x 4 N-subtiles = 8 WMMA accumulators.
// K = (ky*3+kx)*Cin + ci  (matches HWIO weight flattening).

#define AS_STRIDE 40        // 32 k-cols + pad (ushorts)

template <bool UPSAMPLE>
__global__ __launch_bounds__(256)
void conv_wmma(const unsigned short* __restrict__ inp,   // bf16 [B,Cin,Hs,Hs]
               const unsigned short* __restrict__ wpk,   // bf16, prepacked chunks (8KB each)
               float* __restrict__ out,                   // f32  [B,128,128,128]
               int Cin, int Ktot, int Hs) {
    constexpr int H = 128;
    __shared__ __align__(16) unsigned short As[2][128 * AS_STRIDE];
    __shared__ __align__(16) unsigned short Bs[2][16 * 128 * 2];   // 8KB per buffer

    const int tx = blockIdx.x, ty = blockIdx.y, b = blockIdx.z;
    const int t = threadIdx.x;
    const int lane = t & 31, wv = t >> 5;
    const int mPair = wv & 3, nHalf = wv >> 2;
    const int half = lane >> 4, l15 = lane & 15;
    const int oy_base = ty * 8, ox_base = tx * 16;

    v8f acc[2][4] = {};

    // staging thread mappings
    const int kcolA = t >> 3;          // 0..31
    const int pyA   = t & 7;           // tile row (A staging: one 16-px row)

    const int nch = Ktot >> 5;

    auto stageA = [&](int chunk, int buf) {
        const int k0 = chunk << 5;
        const int tap = k0 / Cin;
        const int cbase = k0 - tap * Cin;
        const int ky = tap / 3, kx = tap - ky * 3;
        const int ci = cbase + kcolA;
        const int iy = oy_base + pyA + ky - 1;
        const bool vy = (iy >= 0) && (iy < H);
        const int syi = UPSAMPLE ? (iy >> 1) : iy;
        const unsigned short* src =
            inp + ((size_t)(b * Cin + ci) * Hs + (vy ? syi : 0)) * Hs;
        unsigned short* dst = &As[buf][(pyA * 16) * AS_STRIDE + kcolA];
#pragma unroll
        for (int j = 0; j < 16; ++j) {
            const int ix = ox_base + j + kx - 1;
            unsigned short val = 0;
            if (vy && ix >= 0 && ix < H)
                val = src[UPSAMPLE ? (ix >> 1) : ix];
            dst[j * AS_STRIDE] = val;
        }
    };

    // B staging: one contiguous 8KB chunk -> LDS (async DMA when available)
    auto stageB = [&](int chunk, int buf) {
        const char* g = (const char*)wpk + (size_t)chunk * 8192 + t * 32;
        char* l = (char*)&Bs[buf][0] + t * 32;
#if defined(HAVE_ASYNC_LDS)
        __builtin_amdgcn_global_load_async_to_lds_b128(
            (gv4i_p)(g),      (lv4i_p)(l),      0, 0);
        __builtin_amdgcn_global_load_async_to_lds_b128(
            (gv4i_p)(g + 16), (lv4i_p)(l + 16), 0, 0);
#else
        const uint4* gs = (const uint4*)g;
        uint4* ls = (uint4*)l;
        ls[0] = gs[0];
        ls[1] = gs[1];
#endif
    };

    auto compute = [&](int buf) {
        v16bf afrag[2];
#pragma unroll
        for (int s = 0; s < 2; ++s) {
            const int mrow = (mPair * 2 + s) * 16 + l15;
            v8u au;
#pragma unroll
            for (int i = 0; i < 8; ++i) {
                const int kk = half * 16 + i * 2;
                au[i] = *(const unsigned int*)&As[buf][mrow * AS_STRIDE + kk];
            }
            afrag[s] = __builtin_bit_cast(v16bf, au);
        }
#pragma unroll
        for (int j = 0; j < 4; ++j) {
            const int nloc = nHalf * 64 + j * 16 + l15;
            const int nrot = (nloc + half * 32) & 127;   // undo prepack bank swizzle
            v8u bu;
#pragma unroll
            for (int i = 0; i < 8; ++i) {
                const int kkp = half * 8 + i;
                bu[i] = *(const unsigned int*)&Bs[buf][(kkp * 128 + nrot) * 2];
            }
            const v16bf bfrag = __builtin_bit_cast(v16bf, bu);
#pragma unroll
            for (int s = 0; s < 2; ++s) {
                acc[s][j] = __builtin_amdgcn_wmma_f32_16x16x32_bf16(
                    false, afrag[s], false, bfrag, (short)0, acc[s][j], false, false);
            }
        }
    };

    // prologue: stage chunk 0 into buffer 0
    stageA(0, 0);
    stageB(0, 0);
    wait_async0();
    __syncthreads();

    for (int c = 0; c < nch; ++c) {
        const int cur = c & 1;
        if (c + 1 < nch) {          // overlap: stage next chunk into other buffer
            stageA(c + 1, cur ^ 1);
            stageB(c + 1, cur ^ 1);
        }
        compute(cur);
        wait_async0();              // own async DMAs for c+1 complete
        __syncthreads();            // staging(c+1) visible; reads of buf(cur) done
    }

    // ---- store D per 32-bit C/D layout ----
#pragma unroll
    for (int s = 0; s < 2; ++s) {
#pragma unroll
        for (int j = 0; j < 4; ++j) {
            const int col = nHalf * 64 + j * 16 + l15;
#pragma unroll
            for (int r = 0; r < 8; ++r) {
                const int m = (mPair * 2 + s) * 16 + half * 8 + r;
                const int oy = oy_base + (m >> 4), ox = ox_base + (m & 15);
                out[(((size_t)b * 128 + col) * H + oy) * H + ox] = acc[s][j][r];
            }
        }
    }
}

// ---------- fused (blur) + noise + bias + lrelu + instance-norm + style ----------
template <bool BLUR, bool OUTBF>
__global__ __launch_bounds__(256)
void epilogue_kernel(const float* __restrict__ in,     // [B,C,H,H]
                     const float* __restrict__ noise,  // [B,1,H,H]
                     const float* __restrict__ nw,
                     const float* __restrict__ bias,
                     const float* __restrict__ style,  // [B,2C]
                     void* __restrict__ outp) {
    constexpr int H = 128;
    const int bc = blockIdx.x;
    const int b = bc >> 7, c = bc & 127;
    const float* plane = in + (size_t)bc * H * H;
    const float* np = noise + (size_t)b * H * H;
    const float nwc = nw[c], bb = bias[c];
    const float kw[3] = {0.25f, 0.5f, 0.25f};
    const int t = threadIdx.x;

    float z[64];
    float s = 0.f, sq = 0.f;
#pragma unroll 4
    for (int i = 0; i < 64; ++i) {
        const int p = i * 256 + t;
        const int py = p >> 7, px = p & 127;
        float v;
        if (BLUR) {
            v = 0.f;
#pragma unroll
            for (int dy = -1; dy <= 1; ++dy) {
                const int yy = py + dy;
                if (yy < 0 || yy >= H) continue;
#pragma unroll
                for (int dx = -1; dx <= 1; ++dx) {
                    const int xx = px + dx;
                    if (xx < 0 || xx >= H) continue;
                    v += plane[yy * H + xx] * (kw[dy + 1] * kw[dx + 1]);
                }
            }
        } else {
            v = plane[p];
        }
        v += np[p] * nwc + bb;
        v = v > 0.f ? v : 0.2f * v;   // leaky relu
        z[i] = v;
        s += v;
        sq += v * v;
    }

    __shared__ float rs[256], rq[256];
    rs[t] = s; rq[t] = sq;
    __syncthreads();
    for (int st = 128; st > 0; st >>= 1) {
        if (t < st) { rs[t] += rs[t + st]; rq[t] += rq[t + st]; }
        __syncthreads();
    }
    const float mean = rs[0] * (1.0f / 16384.0f);
    const float var  = rq[0] * (1.0f / 16384.0f) - mean * mean;
    const float inv  = rsqrtf(var + 1e-8f);
    const float s0 = style[b * 256 + c] + 1.0f;
    const float s1 = style[b * 256 + 128 + c];

#pragma unroll 4
    for (int i = 0; i < 64; ++i) {
        const int p = i * 256 + t;
        const float o = (z[i] - mean) * inv * s0 + s1;
        if (OUTBF)
            ((unsigned short*)outp)[(size_t)bc * H * H + p] = f2bf(o);
        else
            ((float*)outp)[(size_t)bc * H * H + p] = o;
    }
}

// ---------- host ----------
extern "C" void kernel_launch(void* const* d_in, const int* in_sizes, int n_in,
                              void* d_out, int out_size, void* d_ws, size_t ws_size,
                              hipStream_t stream) {
    (void)in_sizes; (void)n_in; (void)out_size; (void)ws_size;
    const float* x      = (const float*)d_in[0];   // [8,256,64,64]
    const float* styles = (const float*)d_in[1];   // [8,2,512]
    const float* w_up   = (const float*)d_in[2];   // [3,3,256,128]
    const float* noise1 = (const float*)d_in[3];   // [8,1,128,128]
    const float* nw1    = (const float*)d_in[4];
    const float* b1     = (const float*)d_in[5];
    const float* sw1    = (const float*)d_in[6];   // [512,256]
    const float* sb1    = (const float*)d_in[7];
    const float* w_c1   = (const float*)d_in[8];   // [3,3,128,128]
    const float* noise2 = (const float*)d_in[9];
    const float* nw2    = (const float*)d_in[10];
    const float* b2     = (const float*)d_in[11];
    const float* sw2    = (const float*)d_in[12];
    const float* sb2    = (const float*)d_in[13];
    float* out = (float*)d_out;                    // [8,128,128,128]

    // workspace carve-up
    char* ws = (char*)d_ws;
    size_t off = 0;
    auto carve = [&](size_t bytes) -> char* {
        char* p = ws + off;
        off = (off + bytes + 255) & ~(size_t)255;
        return p;
    };
    const int NX   = 8 * 256 * 64 * 64;        // x elements
    const int NW1  = 2304 * 128;
    const int NW2  = 1152 * 128;
    const int NT   = 8 * 128 * 128 * 128;      // per-tensor elements at 128^2

    unsigned short* xb   = (unsigned short*)carve((size_t)NX * 2);
    unsigned short* w1b  = (unsigned short*)carve((size_t)NW1 * 2);
    unsigned short* w2b  = (unsigned short*)carve((size_t)NW2 * 2);
    float*          st1  = (float*)carve(8 * 256 * 4);
    float*          st2  = (float*)carve(8 * 256 * 4);
    float*          t1   = (float*)carve((size_t)NT * 4);   // conv1 out; reused as conv2 out
    unsigned short* t2b  = (unsigned short*)carve((size_t)NT * 2);
    float*          t3   = t1;                               // alias: t1 dead after epilogue1

    const float wscale1 = sqrtf(2.0f) / sqrtf(2304.0f);     // gain/sqrt(fan_in)
    const float wscale2 = sqrtf(2.0f) / sqrtf(1152.0f);

    // prep conversions
    cvt_bf16_kernel<<<(NX + 255) / 256, 256, 0, stream>>>(x, xb, NX, 1.0f);
    cvt_pack_w_kernel<<<(NW1 + 255) / 256, 256, 0, stream>>>(w_up, w1b, NW1, wscale1);
    cvt_pack_w_kernel<<<(NW2 + 255) / 256, 256, 0, stream>>>(w_c1, w2b, NW2, wscale2);

    // styles
    style_kernel<<<dim3(8, 2), 256, 0, stream>>>(styles, sw1, sb1, sw2, sb2, st1, st2);

    // conv1: upsample(2x) + 3x3, 256 -> 128 at 128^2 (M tile 16x8)
    conv_wmma<true><<<dim3(8, 16, 8), 256, 0, stream>>>(xb, w1b, t1, 256, 2304, 64);

    // blur + epilogue1 -> bf16 activations for conv2
    epilogue_kernel<true, true><<<1024, 256, 0, stream>>>(t1, noise1, nw1, b1, st1, (void*)t2b);

    // conv2: 3x3, 128 -> 128 at 128^2
    conv_wmma<false><<<dim3(8, 16, 8), 256, 0, stream>>>(t2b, w2b, t3, 128, 1152, 128);

    // epilogue2 -> final f32 output
    epilogue_kernel<false, false><<<1024, 256, 0, stream>>>(t3, noise2, nw2, b2, st2, (void*)out);
}
